// GCN_58506044506597
// MI455X (gfx1250) — compile-verified, matrix operand order checked by reading, not run on hardware
//
#include <hip/hip_runtime.h>
#include <hip/hip_bf16.h>
#include <math.h>

typedef __attribute__((ext_vector_type(16))) _Float16 v16h;
typedef __attribute__((ext_vector_type(8)))  float    v8f;

// ---------------------------------------------------------------- utilities

__global__ void zero_f32(float* __restrict__ p, long n) {
    long i = (long)blockIdx.x * blockDim.x + threadIdx.x;
    long stride = (long)gridDim.x * blockDim.x;
    for (; i < n; i += stride) p[i] = 0.0f;
}

// deg[dst[e]] += ew[e]
__global__ void deg_kernel(const int* __restrict__ dst, const float* __restrict__ ew,
                           float* __restrict__ deg, int E) {
    int e = blockIdx.x * blockDim.x + threadIdx.x;
    if (e < E) atomicAdd(&deg[dst[e]], ew[e]);
}

// in-place: deg -> rsqrt(deg + 1)   (deg+1 >= 1 > 0, so no guard needed)
__global__ void dinv_kernel(float* __restrict__ deg, int n) {
    int i = blockIdx.x * blockDim.x + threadIdx.x;
    if (i < n) deg[i] = rsqrtf(deg[i] + 1.0f);
}

// -------------------------------------------------- weight packing (f32->f16
// fragment layout).  Packed index = ((kt*NT + nt)*32 + lane)*16 + e, where
// element e of the v16h maps to k_local = (e<8 ? e : e+8) + (lane&16 ? 8 : 0)
// per the ISA's 16-bit 16x32 fragment layout, and n = nt*16 + (lane&15).
__global__ void pack_w(const float* __restrict__ W, _Float16* __restrict__ Wp,
                       int K, int Nreal, int NT) {
    int idx = blockIdx.x * blockDim.x + threadIdx.x;
    int KT = K >> 5;
    int total = KT * NT * 32;
    if (idx >= total) return;
    int lane = idx & 31;
    int nt   = (idx >> 5) % NT;
    int kt   = (idx >> 5) / NT;
    int n    = (nt << 4) + (lane & 15);
    int kbase = (kt << 5) + ((lane & 16) ? 8 : 0);
    _Float16* o = Wp + (size_t)idx * 16;
#pragma unroll
    for (int e = 0; e < 16; ++e) {
        int k = kbase + (e < 8 ? e : e + 8);
        float v = (n < Nreal) ? W[(size_t)k * Nreal + n] : 0.0f;
        o[e] = (_Float16)v;
    }
}

// -------------------------------------------------------------- WMMA GEMM
// C[M,Nreal] = A[M,K] x B.  One wave owns one 16-row M-tile and ALL NT
// column tiles: the (expensive, f32) A fragment is loaded once per k-step and
// reused for NT back-to-back WMMAs against the tiny L2-resident packed B.
// This cuts A read traffic NT-fold vs one-tile-per-wave (layer 1: 819 MB ->
// 102 MB).  K multiple of 32, M multiple of 16; all control flow before the
// WMMAs is wave-uniform so EXEC is all-ones as required.
template <int NT>
__global__ void __launch_bounds__(128)
wmma_gemm(const float* __restrict__ A, const _Float16* __restrict__ Bp,
          float* __restrict__ C, int M, int K, int Nreal) {
    int lane = threadIdx.x & 31;
    int wave = threadIdx.x >> 5;
    int KT = K >> 5;
    int mt = blockIdx.x * 4 + wave;
    if (mt >= (M >> 4)) return;

    int row = (mt << 4) + (lane & 15);
    int klo = (lane & 16) ? 8 : 0;
    const float* arow = A + (size_t)row * K;
    const v16h* bfrag = (const v16h*)Bp + lane;   // + (kt*NT + nt)*32

    v8f acc[NT];
#pragma unroll
    for (int t = 0; t < NT; ++t) acc[t] = {};

    for (int kt = 0; kt < KT; ++kt) {
        int k0 = (kt << 5) + klo;
        float4 f0 = *(const float4*)(arow + k0);
        float4 f1 = *(const float4*)(arow + k0 + 4);
        float4 f2 = *(const float4*)(arow + k0 + 16);
        float4 f3 = *(const float4*)(arow + k0 + 20);
        v16h a;
        a[0]  = (_Float16)f0.x; a[1]  = (_Float16)f0.y;
        a[2]  = (_Float16)f0.z; a[3]  = (_Float16)f0.w;
        a[4]  = (_Float16)f1.x; a[5]  = (_Float16)f1.y;
        a[6]  = (_Float16)f1.z; a[7]  = (_Float16)f1.w;
        a[8]  = (_Float16)f2.x; a[9]  = (_Float16)f2.y;
        a[10] = (_Float16)f2.z; a[11] = (_Float16)f2.w;
        a[12] = (_Float16)f3.x; a[13] = (_Float16)f3.y;
        a[14] = (_Float16)f3.z; a[15] = (_Float16)f3.w;

        const v16h* bk = bfrag + (size_t)kt * NT * 32;
#pragma unroll
        for (int nt = 0; nt < NT; ++nt) {
            v16h b = bk[nt * 32];
            acc[nt] = __builtin_amdgcn_wmma_f32_16x16x32_f16(
                /*neg_a=*/false, a, /*neg_b=*/false, b,
                /*c_mod=*/(short)0, acc[nt],
                /*reuse_a=*/false, /*reuse_b=*/false);
        }
    }

    int r0 = (mt << 4) + ((lane & 16) ? 8 : 0);
#pragma unroll
    for (int nt = 0; nt < NT; ++nt) {
        int col = (nt << 4) + (lane & 15);
        if (col < Nreal) {
#pragma unroll
            for (int v = 0; v < 8; ++v)
                C[(size_t)(r0 + v) * Nreal + col] = acc[nt][v];
        }
    }
}

// ----------------------------------------------------- edge scatter-gather
// One wave per edge; lane strides over features.  agg and xw are L2-resident
// (51 MB + 51 MB < 192 MB L2) so the random 512B gathers/atomics stay on-chip.
__global__ void scatter_kernel(const int* __restrict__ src, const int* __restrict__ dst,
                               const float* __restrict__ ew, const float* __restrict__ dinv,
                               const float* __restrict__ XW, float* __restrict__ AGG,
                               int E, int F) {
    int gid  = blockIdx.x * blockDim.x + threadIdx.x;
    int e    = gid >> 5;
    int lane = gid & 31;
    if (e >= E) return;
    int s = src[e], d = dst[e];
    float nrm = dinv[s] * ew[e] * dinv[d];
    const float* xr = XW + (size_t)s * F;
    float* ar = AGG + (size_t)d * F;
    for (int f = lane; f < F; f += 32)
        atomicAdd(&ar[f], nrm * xr[f]);
}

// self-loop + bias + ReLU, in place on agg (-> h)
__global__ void finalize1(float* __restrict__ agg, const float* __restrict__ xw,
                          const float* __restrict__ dinv, const float* __restrict__ b,
                          int n, int F) {
    long i = (long)blockIdx.x * blockDim.x + threadIdx.x;
    if (i >= (long)n * F) return;
    int row = (int)(i / F);
    int f   = (int)(i % F);
    float dd = dinv[row];
    float v = agg[i] + dd * dd * xw[i] + b[f];
    agg[i] = v > 0.0f ? v : 0.0f;
}

// self-loop + bias, then row L2-normalize (one wave per node, F=40)
__global__ void finalize2(const float* __restrict__ agg, const float* __restrict__ xw,
                          const float* __restrict__ dinv, const float* __restrict__ b,
                          float* __restrict__ out, int n) {
    int wid  = (blockIdx.x * blockDim.x + threadIdx.x) >> 5;
    int lane = threadIdx.x & 31;
    if (wid >= n) return;
    const int F = 40;
    float dd = dinv[wid]; dd *= dd;
    long base = (long)wid * F;
    float v0 = agg[base + lane] + dd * xw[base + lane] + b[lane];
    float v1 = 0.0f;
    if (lane < 8)
        v1 = agg[base + 32 + lane] + dd * xw[base + 32 + lane] + b[32 + lane];
    float ss = v0 * v0 + v1 * v1;
#pragma unroll
    for (int m = 16; m >= 1; m >>= 1) ss += __shfl_xor(ss, m, 32);
    float scale = 1.0f / fmaxf(sqrtf(ss), 1e-12f);
    out[base + lane] = v0 * scale;
    if (lane < 8) out[base + 32 + lane] = v1 * scale;
}

// ------------------------------------------------------------------- launch

extern "C" void kernel_launch(void* const* d_in, const int* in_sizes, int n_in,
                              void* d_out, int out_size, void* d_ws, size_t ws_size,
                              hipStream_t stream) {
    const float* x  = (const float*)d_in[0];
    const int*   ei = (const int*)d_in[1];
    const float* ew = (const float*)d_in[2];
    const float* W1 = (const float*)d_in[3];
    const float* b1 = (const float*)d_in[4];
    const float* W2 = (const float*)d_in[5];
    const float* b2 = (const float*)d_in[6];
    float* out = (float*)d_out;

    const int NFEAT = 256, NHID = 128, NCLASS = 40;
    const int n = in_sizes[0] / NFEAT;       // 100000 (multiple of 16)
    const int E = in_sizes[2];               // 1600000
    const int* src = ei;
    const int* dst = ei + E;

    // workspace layout
    char* ws = (char*)d_ws;
    size_t off = 0;
    auto walloc = [&](size_t bytes) -> char* {
        char* p = ws + off;
        off += (bytes + 255) & ~(size_t)255;
        return p;
    };
    float*    dinv = (float*)walloc((size_t)n * 4);               // deg -> dinv
    float*    xw1  = (float*)walloc((size_t)n * NHID * 4);
    float*    agg1 = (float*)walloc((size_t)n * NHID * 4);        // becomes h
    float*    xw2  = (float*)walloc((size_t)n * NCLASS * 4);
    float*    agg2 = (float*)walloc((size_t)n * NCLASS * 4);
    _Float16* W1p  = (_Float16*)walloc((size_t)NFEAT * NHID * 2);
    _Float16* W2p  = (_Float16*)walloc((size_t)NHID * 48 * 2);    // padded to 48 cols

    // zero accumulators (ws is poisoned, not zeroed, by the harness)
    zero_f32<<<1024, 256, 0, stream>>>(dinv, n);
    zero_f32<<<4096, 256, 0, stream>>>(agg1, (long)n * NHID);
    zero_f32<<<4096, 256, 0, stream>>>(agg2, (long)n * NCLASS);

    // degree + sym-norm factors
    deg_kernel<<<(E + 255) / 256, 256, 0, stream>>>(dst, ew, dinv, E);
    dinv_kernel<<<(n + 255) / 256, 256, 0, stream>>>(dinv, n);

    // pack weights into WMMA B-fragment layout (f32 -> f16, zero-padded cols)
    const int NT1 = NHID / 16;   // 8
    const int NT2 = 3;           // ceil(40/16)
    pack_w<<<((NFEAT / 32) * NT1 * 32 + 255) / 256, 256, 0, stream>>>(W1, W1p, NFEAT, NHID, NT1);
    pack_w<<<((NHID  / 32) * NT2 * 32 + 255) / 256, 256, 0, stream>>>(W2, W2p, NHID, NCLASS, NT2);

    const int MT = n / 16;                   // 6250 row tiles, 4 waves/block

    // layer 1: xw1 = x @ W1   (WMMA f16 inputs, f32 accumulate)
    wmma_gemm<8><<<(MT + 3) / 4, 128, 0, stream>>>(x, W1p, xw1, n, NFEAT, NHID);

    // layer-1 aggregation + self-loop + bias + ReLU  (agg1 -> h in place)
    {
        long thr = (long)E * 32;
        scatter_kernel<<<(unsigned)((thr + 255) / 256), 256, 0, stream>>>(src, dst, ew, dinv, xw1, agg1, E, NHID);
        finalize1<<<(unsigned)(((long)n * NHID + 255) / 256), 256, 0, stream>>>(agg1, xw1, dinv, b1, n, NHID);
    }

    // layer 2: xw2 = h @ W2
    wmma_gemm<3><<<(MT + 3) / 4, 128, 0, stream>>>(agg1, W2p, xw2, n, NHID, NCLASS);

    // layer-2 aggregation + self-loop + bias + row L2-normalize
    {
        long thr = (long)E * 32;
        scatter_kernel<<<(unsigned)((thr + 255) / 256), 256, 0, stream>>>(src, dst, ew, dinv, xw2, agg2, E, NCLASS);
        finalize2<<<(unsigned)(((long)n * 32 + 255) / 256), 256, 0, stream>>>(agg2, xw2, dinv, b2, out, n);
    }
}
